// myNN_13443247637122
// MI455X (gfx1250) — matrix operand order repaired and checked by reading, not compiled, not run on hardware
//
#include <hip/hip_runtime.h>

// ---------------------------------------------------------------------------
// Sizes (fixed by the reference)
// ---------------------------------------------------------------------------
#define NN    16384   // nodes
#define INF_  128     // in features
#define HID   128     // hidden
#define OUTF  2
#define NG    64      // graphs
#define BM    64      // rows of h per workgroup
#define KB    32      // K-chunk (matches wmma 16x16x32)
#define LDA   40      // ushorts per staged row (32 data + 8 pad -> 80B, bank spread)
#define NITER (NN / KB)

typedef __attribute__((ext_vector_type(16))) __bf16 bf16x16;
typedef __attribute__((ext_vector_type(8)))  float  floatx8;
typedef __attribute__((ext_vector_type(4)))  float  floatx4;   // native vector for NT loads

union FragAB { uint4 q[2]; bf16x16 v; };
union PK4    { __bf16 h[4]; uint2 u; };
union PK1    { __bf16 h; unsigned short s; };

// IEEE-monotonic float atomic max via sign-split int/uint atomics.
__device__ __forceinline__ void atomicMaxF(float* a, float v) {
    unsigned int u = __float_as_uint(v);
    if ((u >> 31) == 0) atomicMax((int*)a, (int)u);
    else                atomicMin((unsigned int*)a, u);
}

// ---------------------------------------------------------------------------
// g <- -inf
// ---------------------------------------------------------------------------
__global__ void ginit_kernel(float* g) {
    int idx = blockIdx.x * blockDim.x + threadIdx.x;   // 8192 cells
    if (idx < NG * HID) g[idx] = __uint_as_float(0xFF800000u);
}

// ---------------------------------------------------------------------------
// supportT[j][n] = bf16( sum_k x[n][k] * W_gcn[k][j] )   (exact f32 math)
// ---------------------------------------------------------------------------
__global__ __launch_bounds__(256) void support_kernel(
        const float* __restrict__ x, const float* __restrict__ Wg,
        unsigned short* __restrict__ supportT) {
    __shared__ float xs[16 * INF_];
    const int tid = threadIdx.x;
    const int rowbase = blockIdx.x * 16;
    #pragma unroll
    for (int i = 0; i < 8; ++i) {
        int idx = tid + i * 256;
        xs[idx] = x[(size_t)rowbase * INF_ + idx];
    }
    __syncthreads();
    const int j = tid & 127;
    const int half = tid >> 7;
    float acc[8] = {0.f,0.f,0.f,0.f,0.f,0.f,0.f,0.f};
    for (int k = 0; k < INF_; ++k) {
        float w = Wg[k * HID + j];
        #pragma unroll
        for (int r = 0; r < 8; ++r) acc[r] += xs[(half * 8 + r) * INF_ + k] * w;
    }
    #pragma unroll
    for (int r = 0; r < 8; ++r) {
        int n = rowbase + half * 8 + r;
        PK1 p; p.h = (__bf16)acc[r];           // native v_cvt (RNE)
        supportT[(size_t)j * NN + n] = p.s;
    }
}

// ---------------------------------------------------------------------------
// Fused, pipelined:  h_tile = leaky_relu(adj_tile @ support + b) -> seg-max
// Manual x2 unroll: all LDS/register buffer indices are literals (no movrel).
// adj prefetched a full iteration ahead into alternating reg sets (NT loads),
// support tile DMA'd with global_load_async_to_lds_b128 (ASYNCcnt).
// One barrier per K-iteration.
// ---------------------------------------------------------------------------
struct __align__(16) SMemT {
    union U {
        struct {
            unsigned short A[2][BM * LDA];    // 2 x 5120 B  (bf16 adj tiles)
            unsigned short B[2][HID * LDA];   // 2 x 10240 B (bf16 support tiles)
        } s;                                  // 30720 B total
        float h[BM * 129];                    // 33024 B (post-GEMM tile)
    } u;
    int sb[BM];
};

__global__ __launch_bounds__(256) void gcn_wmma_kernel(
        const float* __restrict__ adj,
        const unsigned short* __restrict__ supportT,   // [HID][NN] bf16
        const float* __restrict__ b_gcn,
        const int* __restrict__ batch,
        float* __restrict__ g) {                       // [NG][HID]
    __shared__ SMemT sm;
    const int tid  = threadIdx.x;
    const int wave = tid >> 5;
    const int lane = tid & 31;
    const int rowbase = blockIdx.x * BM;

    if (tid < BM) sm.sb[tid] = batch[rowbase + tid];

    const int rt  = wave >> 1;         // row tile 0..3
    const int cg  = wave & 1;          // column half 0..1
    const int l16 = lane & 15;
    const int hi  = lane >> 4;

    // cooperative-load geometry
    const int arow = tid >> 3;                 // 0..31 (x2 passes)
    const int ac   = (tid & 7) * 4;            // K offset, float units

    floatx8 acc[4];
    floatx8 zero = {0.f,0.f,0.f,0.f,0.f,0.f,0.f,0.f};
    #pragma unroll
    for (int i = 0; i < 4; ++i) acc[i] = zero;

    floatx4 ra[2], rb[2];                      // two named prefetch reg sets

    auto loadA = [&](int k0, floatx4* dst) {
        #pragma unroll
        for (int p = 0; p < 2; ++p)
            dst[p] = __builtin_nontemporal_load(
                reinterpret_cast<const floatx4*>(
                    adj + (size_t)(rowbase + p * 32 + arow) * NN + k0 + ac));
    };
    auto storeA = [&](int buf, const floatx4* src) {
        #pragma unroll
        for (int p = 0; p < 2; ++p) {
            PK4 pk;                              // hardware packed cvt, RNE
            pk.h[0] = (__bf16)src[p].x;
            pk.h[1] = (__bf16)src[p].y;
            pk.h[2] = (__bf16)src[p].z;
            pk.h[3] = (__bf16)src[p].w;
            *reinterpret_cast<uint2*>(&sm.u.s.A[buf][(p * 32 + arow) * LDA + ac]) = pk.u;
        }
    };
    auto asyncB = [&](int k0, int buf) {
        #pragma unroll
        for (int p = 0; p < 2; ++p) {
            int chunk = tid + p * 256;         // 512 x 16B chunks
            int col = chunk >> 2;
            int co  = (chunk & 3) * 8;         // ushort offset
            const unsigned short* gsrc = supportT + (size_t)col * NN + k0 + co;
            unsigned ldst = (unsigned)(size_t)(&sm.u.s.B[buf][col * LDA + co]);
            asm volatile("global_load_async_to_lds_b128 %0, %1, off"
                         :: "v"(ldst), "v"(gsrc) : "memory");
        }
    };
    auto compute = [&](int c) {                // c is always a literal
        FragAB fa;
        const char* ab = (const char*)sm.u.s.A[c]
                       + (rt * 16 + l16) * (LDA * 2) + hi * 16;
        fa.q[0] = *reinterpret_cast<const uint4*>(ab);
        fa.q[1] = *reinterpret_cast<const uint4*>(ab + 32);
        #pragma unroll
        for (int nn = 0; nn < 4; ++nn) {
            FragAB fb;
            const int col = (cg * 4 + nn) * 16 + l16;
            const char* bb = (const char*)sm.u.s.B[c]
                           + col * (LDA * 2) + hi * 16;
            fb.q[0] = *reinterpret_cast<const uint4*>(bb);
            fb.q[1] = *reinterpret_cast<const uint4*>(bb + 32);
            acc[nn] = __builtin_amdgcn_wmma_f32_16x16x32_bf16(
                false, fa.v, false, fb.v, (short)0, acc[nn], false, false);
        }
    };

    // ---- prologue: chunk0 -> LDS buf0 (via ra), chunk1 -> reg set rb
    loadA(0, ra);
    asyncB(0, 0);
    storeA(0, ra);
    loadA(KB, rb);
    asm volatile("s_wait_asynccnt 0x0" ::: "memory");

    // ---- main loop, manually unrolled x2 (NITER is even)
    for (int it = 0; it < NITER; it += 2) {
        // === even body: consume buf0; fill buf1 from rb; prefetch into ra ===
        __syncthreads();
        asyncB((it + 1) * KB, 1);
        storeA(1, rb);                          // chunk it+1
        if (it + 2 < NITER) loadA((it + 2) * KB, ra);
        compute(0);
        asm volatile("s_wait_asynccnt 0x0" ::: "memory");

        // === odd body: consume buf1; fill buf0 from ra; prefetch into rb ===
        __syncthreads();
        if (it + 2 < NITER) {
            asyncB((it + 2) * KB, 0);
            storeA(0, ra);                      // chunk it+2
            loadA((it + 3) * KB, rb);           // it+3 <= 511, always valid here
        }
        compute(1);
        if (it + 2 < NITER)
            asm volatile("s_wait_asynccnt 0x0" ::: "memory");
    }
    __syncthreads();

    // ---- epilogue: bias + leaky_relu into LDS tile (overlays staging)
    #pragma unroll
    for (int nn = 0; nn < 4; ++nn) {
        const int coln = (cg * 4 + nn) * 16 + l16;
        const float bias = b_gcn[coln];
        const int mbase = rt * 16 + hi * 8;
        #pragma unroll
        for (int r = 0; r < 8; ++r) {
            float v = acc[nn][r] + bias;
            v = v > 0.f ? v : 0.01f * v;        // leaky_relu(0.01)
            sm.u.h[(mbase + r) * 129 + coln] = v;
        }
    }
    __syncthreads();

    // ---- run-compressed segment max (batch sorted); atomics merge tiles
    if (tid < HID) {
        int cur = sm.sb[0];
        float m = __uint_as_float(0xFF800000u);
        for (int r = 0; r < BM; ++r) {
            int b = sm.sb[r];
            if (b != cur) {
                atomicMaxF(&g[cur * HID + tid], m);
                cur = b;
                m = __uint_as_float(0xFF800000u);
            }
            m = fmaxf(m, sm.u.h[r * 129 + tid]);
        }
        atomicMaxF(&g[cur * HID + tid], m);
    }
}

// ---------------------------------------------------------------------------
// MLP head (tiny, exact f32):  out = lrelu?(in @ W.T + b)
// ---------------------------------------------------------------------------
__global__ void lin_kernel(const float* __restrict__ in, const float* __restrict__ W,
                           const float* __restrict__ b, float* __restrict__ out,
                           int relu) {
    int idx = blockIdx.x * blockDim.x + threadIdx.x;
    int i = idx >> 7, j = idx & 127;
    const float* gi = in + i * HID;
    const float* wj = W + j * HID;
    float a = b[j];
    #pragma unroll 4
    for (int k = 0; k < HID; ++k) a += gi[k] * wj[k];
    if (relu) a = a > 0.f ? a : 0.01f * a;
    out[idx] = a;
}

__global__ void out_kernel(const float* __restrict__ g2, const float* __restrict__ Wo,
                           const float* __restrict__ bo, float* __restrict__ out) {
    int idx = blockIdx.x * blockDim.x + threadIdx.x;
    if (idx >= NG * OUTF) return;
    int i = idx >> 1, o = idx & 1;
    float a = bo[o];
    #pragma unroll 4
    for (int k = 0; k < HID; ++k) a += g2[i * HID + k] * Wo[o * HID + k];
    out[idx] = a;
}

// ---------------------------------------------------------------------------
extern "C" void kernel_launch(void* const* d_in, const int* in_sizes, int n_in,
                              void* d_out, int out_size, void* d_ws, size_t ws_size,
                              hipStream_t stream) {
    const float* x     = (const float*)d_in[0];
    const float* adj   = (const float*)d_in[1];
    const float* W_gcn = (const float*)d_in[2];
    const float* b_gcn = (const float*)d_in[3];
    const float* W_l1  = (const float*)d_in[4];
    const float* b_l1  = (const float*)d_in[5];
    const float* W_l2  = (const float*)d_in[6];
    const float* b_l2  = (const float*)d_in[7];
    const float* W_out = (const float*)d_in[8];
    const float* b_out = (const float*)d_in[9];
    const int*   batch = (const int*)d_in[10];
    // d_in[11] n_nodes unused (G fixed at 64)

    char* ws = (char*)d_ws;
    unsigned short* supportT = (unsigned short*)ws;            // 4 MiB bf16
    float* g  = (float*)(ws + (size_t)HID * NN * 2);
    float* g1 = g  + NG * HID;
    float* g2 = g1 + NG * HID;
    float* out = (float*)d_out;

    ginit_kernel<<<32, 256, 0, stream>>>(g);
    support_kernel<<<NN / 16, 256, 0, stream>>>(x, W_gcn, supportT);
    gcn_wmma_kernel<<<NN / BM, 256, 0, stream>>>(adj, supportT, b_gcn, batch, g);
    lin_kernel<<<(NG * HID) / 256, 256, 0, stream>>>(g,  W_l1, b_l1, g1, 1);
    lin_kernel<<<(NG * HID) / 256, 256, 0, stream>>>(g1, W_l2, b_l2, g2, 1);
    out_kernel<<<1, 128, 0, stream>>>(g2, W_out, b_out, out);
}
